// LSTMWrapper_69054484185609
// MI455X (gfx1250) — compile-verified, matrix operand order checked by reading, not run on hardware
//
#include <hip/hip_runtime.h>

// ---------------------------------------------------------------------------
// LSTM  (B=32, T=2048, D=256, U=256)  for MI455X / gfx1250 (wave32, WMMA)
//
// Phase 0: convert data -> bf16; swizzle Wx/Wh into WMMA-B fragment layout;
//          zero h double-buffer; reset grid-barrier counter.
// Phase 1: zx = data @ Wx + b   (bf16 WMMA, f32 accumulate, f32 zx in ws)
//          A tile staged into LDS with GLOBAL_LOAD_ASYNC_TO_LDS_B128.
// Phase 2: persistent scan: 32 single-wave WGs, Wh slice in LDS (async-
//          staged), c in VGPRs, h exchanged via L2-resident bf16 buffer,
//          atomic phase barrier per step, zx prefetched one step ahead.
// ---------------------------------------------------------------------------

typedef __bf16 v16bf __attribute__((ext_vector_type(16)));
typedef float  v8f   __attribute__((ext_vector_type(8)));
typedef int    v4i   __attribute__((ext_vector_type(4)));
typedef unsigned short v4u16 __attribute__((ext_vector_type(4)));

struct Frag { v4i lo; v4i hi; };   // 32 bytes == v16bf

#define B_  32
#define T_  2048
#define D_  256
#define U_  256
#define G4  1024   // 4*U

// ---- workspace layout (bytes) ---------------------------------------------
static constexpr size_t ZX_OFF     = 0;                                 // f32 [B*T, 4U]
static constexpr size_t ZX_BYTES   = (size_t)B_ * T_ * G4 * 4;          // 256 MB
static constexpr size_t XB_OFF     = ZX_OFF + ZX_BYTES;                 // bf16 [B*T, D]
static constexpr size_t XB_BYTES   = (size_t)B_ * T_ * D_ * 2;          // 32 MB
static constexpr size_t WXB_OFF    = XB_OFF + XB_BYTES;                 // bf16 swizzled Wx
static constexpr size_t WB_BYTES   = (size_t)D_ * G4 * 2;               // 512 KB
static constexpr size_t WHB_OFF    = WXB_OFF + WB_BYTES;                // bf16 swizzled Wh
static constexpr size_t HBUF_OFF   = WHB_OFF + WB_BYTES;                // bf16 h x2
static constexpr unsigned HBUF_ELEMS = B_ * U_;                         // 8192
static constexpr size_t CTR_OFF    = HBUF_OFF + 2 * (size_t)HBUF_ELEMS * 2;

__device__ __forceinline__ unsigned short f2bf(float f) {
  return __builtin_bit_cast(unsigned short, (__bf16)f);
}
__device__ __forceinline__ float sigf(float x) {
  return 1.0f / (1.0f + __expf(-x));
}
__device__ __forceinline__ float tanh_fast(float x) {
  return 1.0f - 2.0f / (1.0f + __expf(2.0f * x));
}

// Async global->LDS copy, 2 x b128 per lane (32B), CDNA5 TDM-adjacent path.
// INST_OFFSET is applied to BOTH the LDS and global address (ISA 08 §4.4),
// so one (lds_off, gl_off) pair covers the offset:0 / offset:16 pair.
__device__ __forceinline__ void async_copy_32B(unsigned lds_off, unsigned gl_off,
                                               const void* base) {
  asm volatile("global_load_async_to_lds_b128 %0, %1, %2 offset:0"
               :: "v"(lds_off), "v"(gl_off), "s"(base) : "memory");
  asm volatile("global_load_async_to_lds_b128 %0, %1, %2 offset:16"
               :: "v"(lds_off), "v"(gl_off), "s"(base) : "memory");
}
__device__ __forceinline__ void wait_async0() {
  asm volatile("s_wait_asynccnt 0x0" ::: "memory");
}

// ---- Phase 0a: data (f32) -> bf16, row-major ------------------------------
__global__ __launch_bounds__(256) void k_cvt_data(const float* __restrict__ x,
                                                  unsigned short* __restrict__ xb) {
  const size_t i = ((size_t)blockIdx.x * blockDim.x + threadIdx.x) * 4;
  const float4 v = *(const float4*)(x + i);
  v4u16 o;
  o.x = f2bf(v.x); o.y = f2bf(v.y); o.z = f2bf(v.z); o.w = f2bf(v.w);
  *(v4u16*)(xb + i) = o;
}

// ---- Phase 0b: swizzle W into WMMA-B fragment slabs, init state -----------
// B-fragment (bf16 K=32 x N=16): lane L holds column n = nt*16 + (L&15),
// K-base kb = (L>>4)*8; elems 0..7 -> K kb..kb+7, elems 8..15 -> K kb+16..23.
// Stored as 16 contiguous bf16 (32 B) per (ntile, kchunk, lane).
__global__ __launch_bounds__(256) void k_prep(const float* __restrict__ Wx,
                                              const float* __restrict__ Wh,
                                              unsigned short* __restrict__ wxb,
                                              unsigned short* __restrict__ whb,
                                              unsigned short* __restrict__ hbuf,
                                              unsigned* __restrict__ ctr) {
  const unsigned tid = blockIdx.x * blockDim.x + threadIdx.x;
  if (tid < (unsigned)(D_ * G4)) {
    const unsigned idx  = tid & 15;
    const unsigned lane = (tid >> 4) & 31;
    const unsigned kc   = (tid >> 9) & 7;
    const unsigned nt   = tid >> 12;
    const unsigned k = kc * 32 + (lane >> 4) * 8 + (idx < 8 ? idx : idx + 8);
    const unsigned n = nt * 16 + (lane & 15);
    wxb[tid] = f2bf(Wx[(size_t)k * G4 + n]);
    whb[tid] = f2bf(Wh[(size_t)k * G4 + n]);
  } else if (tid < (unsigned)(D_ * G4) + 2u * HBUF_ELEMS) {
    hbuf[tid - (unsigned)(D_ * G4)] = 0;   // h0 = 0, both buffers
  } else if (tid == (unsigned)(D_ * G4) + 2u * HBUF_ELEMS) {
    *ctr = 0;                              // reset grid barrier each call
  }
}

// ---- Phase 1: zx = data @ Wx + b  -----------------------------------------
// Block = 256 thr = 8 waves; block owns one 16-row tile of [B*T], covers all
// 1024 columns (wave w -> columns [w*128, w*128+128)). A tile async-staged.
__global__ __launch_bounds__(256) void k_gemm_zx(const unsigned short* __restrict__ xb,
                                                 const unsigned short* __restrict__ wxb,
                                                 const float* __restrict__ bias,
                                                 float* __restrict__ zx) {
  __shared__ unsigned short sA[16 * D_];   // 8 KB bf16 A tile
  const unsigned mtile = blockIdx.x;
  const unsigned tid   = threadIdx.x;

  {  // async stage: tile = 16 contiguous rows = 8 KB; 32 B per thread
    const unsigned lds_off = (unsigned)(size_t)(sA) + tid * 32;
    const unsigned gl_off  = tid * 32;
    async_copy_32B(lds_off, gl_off, xb + (size_t)mtile * (16 * D_));
    wait_async0();
  }
  __syncthreads();

  const unsigned w = tid >> 5, lane = tid & 31;
  const unsigned mhalf = lane >> 4, n16 = lane & 15;
  const unsigned kb = mhalf * 8;
  const unsigned am = n16;                 // A row within tile

  v8f acc[8];
  #pragma unroll
  for (int j = 0; j < 8; ++j) { v8f z = {}; acc[j] = z; }

  #pragma unroll
  for (int kc = 0; kc < 8; ++kc) {
    Frag af;
    const unsigned aoff = am * D_ + kc * 32 + kb;
    af.lo = *(const v4i*)(sA + aoff);
    af.hi = *(const v4i*)(sA + aoff + 16);
    const v16bf a = __builtin_bit_cast(v16bf, af);
    #pragma unroll
    for (int j = 0; j < 8; ++j) {
      const unsigned nt = w * 8 + j;
      const v4i* bp = (const v4i*)(wxb + (((size_t)nt * 8 + kc) * 32 + lane) * 16);
      Frag bf_; bf_.lo = bp[0]; bf_.hi = bp[1];
      const v16bf bv = __builtin_bit_cast(v16bf, bf_);
      acc[j] = __builtin_amdgcn_wmma_f32_16x16x32_bf16(
          false, a, false, bv, (short)0, acc[j], false, false);
    }
  }

  #pragma unroll
  for (int j = 0; j < 8; ++j) {
    const unsigned col = w * 128 + j * 16 + n16;
    const float bv = bias[col];
    #pragma unroll
    for (int r = 0; r < 8; ++r) {
      const unsigned row = mtile * 16 + r + mhalf * 8;   // C layout: M=r / r+8
      zx[(size_t)row * G4 + col] = acc[j][r] + bv;
    }
  }
}

// ---- Phase 2: persistent recurrent scan -----------------------------------
// 32 WGs x 1 wave. WG (mt, uc): batch rows [mt*16, mt*16+16), u-cols
// [uc*16, uc*16+16) => z-cols {g*256 + uc*16 ..} for g=i,f,g,o.
__global__ __launch_bounds__(32) void k_lstm_scan(const unsigned short* __restrict__ whb,
                                                  const float* __restrict__ zx,
                                                  unsigned short* __restrict__ hbuf,
                                                  unsigned* __restrict__ ctr,
                                                  float* __restrict__ out) {
  __shared__ unsigned short sB[4 * 8 * 32 * 16];   // 32 KB: [gate][kchunk][lane][16]
  const unsigned lane = threadIdx.x;
  const unsigned mt = blockIdx.x >> 4;
  const unsigned uc = blockIdx.x & 15;
  const unsigned u0 = uc * 16;

  // Async-stage this WG's Wh slice (already in B-fragment layout) into LDS.
  for (int g = 0; g < 4; ++g) {
    const unsigned nt = g * 16 + uc;
    #pragma unroll
    for (int kc = 0; kc < 8; ++kc) {
      const unsigned lds_off =
          (unsigned)(size_t)(sB) + ((g * 8 + kc) * 32 + lane) * 32;
      const unsigned gl_off = (((nt * 8u + kc) * 32u + lane) * 16u) * 2u;
      async_copy_32B(lds_off, gl_off, whb);
    }
  }
  wait_async0();
  __syncthreads();

  const unsigned mhalf = lane >> 4, n16 = lane & 15;
  const unsigned am = mt * 16 + n16;      // A (=h) batch row for this lane
  const unsigned kb = mhalf * 8;

  size_t zoff[8], ooff[8];
  unsigned hoff[8];
  float c[8];
  #pragma unroll
  for (int r = 0; r < 8; ++r) {
    const unsigned mrow = mt * 16 + r + mhalf * 8;      // C layout row
    zoff[r] = (size_t)mrow * T_ * G4 + u0 + n16;
    ooff[r] = (size_t)mrow * T_ * U_ + u0 + n16;
    hoff[r] = mrow * U_ + u0 + n16;
    c[r] = 0.0f;
  }

  for (int t = 0; t < T_; ++t) {
    const unsigned short* rb = hbuf + (unsigned)(t & 1) * HBUF_ELEMS;
    unsigned short* wb = hbuf + (unsigned)((t + 1) & 1) * HBUF_ELEMS;

    // Prefetch next step's zx slices so they overlap this step's WMMA chain.
    {
      const int tn = (t + 1 < T_) ? (t + 1) : t;
      const size_t ztn = (size_t)tn * G4;
      #pragma unroll
      for (int g = 0; g < 4; ++g)
        __builtin_prefetch(&zx[zoff[0] + ztn + (unsigned)g * U_], 0, 1);
    }

    v8f acc[4];
    #pragma unroll
    for (int g = 0; g < 4; ++g) { v8f z = {}; acc[g] = z; }

    #pragma unroll
    for (int kc = 0; kc < 8; ++kc) {
      Frag af;
      const unsigned aoff = am * U_ + kc * 32 + kb;
      af.lo = *(const v4i*)(rb + aoff);          // K kb..kb+7   (16B)
      af.hi = *(const v4i*)(rb + aoff + 16);     // K kb+16..23  (16B)
      const v16bf a = __builtin_bit_cast(v16bf, af);
      #pragma unroll
      for (int g = 0; g < 4; ++g) {
        const v4i* bp = (const v4i*)(sB + ((g * 8 + kc) * 32 + lane) * 16);
        Frag bf_; bf_.lo = bp[0]; bf_.hi = bp[1];
        const v16bf bv = __builtin_bit_cast(v16bf, bf_);
        acc[g] = __builtin_amdgcn_wmma_f32_16x16x32_bf16(
            false, a, false, bv, (short)0, acc[g], false, false);
      }
    }

    const size_t zt = (size_t)t * G4;
    #pragma unroll
    for (int r = 0; r < 8; ++r) {
      const float zi = acc[0][r] + zx[zoff[r] + zt + 0 * U_];
      const float zf = acc[1][r] + zx[zoff[r] + zt + 1 * U_];
      const float zg = acc[2][r] + zx[zoff[r] + zt + 2 * U_];
      const float zo = acc[3][r] + zx[zoff[r] + zt + 3 * U_];
      const float ig = sigf(zi), fg = sigf(zf);
      const float gg = tanh_fast(zg), og = sigf(zo);
      const float cn = fg * c[r] + ig * gg;
      c[r] = cn;
      const float hn = og * tanh_fast(cn);
      out[ooff[r] + (size_t)t * U_] = hn;        // fp32 result sequence
      wb[hoff[r]] = f2bf(hn);                    // bf16 h for next step
    }

    // Grid-wide phase barrier (monotonic counter, release/acquire via L2).
    __builtin_amdgcn_fence(__ATOMIC_RELEASE, "agent");
    if (lane == 0) {
      __hip_atomic_fetch_add(ctr, 1u, __ATOMIC_RELEASE, __HIP_MEMORY_SCOPE_AGENT);
      const unsigned target = 32u * (unsigned)(t + 1);
      while (__hip_atomic_load(ctr, __ATOMIC_ACQUIRE, __HIP_MEMORY_SCOPE_AGENT) < target)
        __builtin_amdgcn_s_sleep(2);
    }
    __builtin_amdgcn_fence(__ATOMIC_ACQUIRE, "agent");
    __builtin_amdgcn_wave_barrier();
  }
}

// ---------------------------------------------------------------------------
extern "C" void kernel_launch(void* const* d_in, const int* in_sizes, int n_in,
                              void* d_out, int out_size, void* d_ws, size_t ws_size,
                              hipStream_t stream) {
  const float* data = (const float*)d_in[0];   // [B, T, D]
  const float* Wx   = (const float*)d_in[1];   // [D, 4U]
  const float* Wh   = (const float*)d_in[2];   // [U, 4U]
  const float* bias = (const float*)d_in[3];   // [4U]
  float* outp = (float*)d_out;                 // [B, T, U]

  char* ws = (char*)d_ws;
  float*          zxp  = (float*)(ws + ZX_OFF);
  unsigned short* xb   = (unsigned short*)(ws + XB_OFF);
  unsigned short* wxb  = (unsigned short*)(ws + WXB_OFF);
  unsigned short* whb  = (unsigned short*)(ws + WHB_OFF);
  unsigned short* hbuf = (unsigned short*)(ws + HBUF_OFF);
  unsigned*       ctr  = (unsigned*)(ws + CTR_OFF);

  // Phase 0: conversions / swizzles / state init
  hipLaunchKernelGGL(k_cvt_data, dim3((B_ * T_ * D_) / (256 * 4)), dim3(256), 0, stream,
                     data, xb);
  hipLaunchKernelGGL(k_prep, dim3((D_ * G4 + 2 * HBUF_ELEMS + 1 + 255) / 256), dim3(256),
                     0, stream, Wx, Wh, wxb, whb, hbuf, ctr);

  // Phase 1: zx = data @ Wx + b   (one 16-row tile per block, 8 waves wide)
  hipLaunchKernelGGL(k_gemm_zx, dim3((B_ * T_) / 16), dim3(256), 0, stream,
                     xb, wxb, bias, zxp);

  // Phase 2: persistent recurrence, 32 single-wave workgroups
  hipLaunchKernelGGL(k_lstm_scan, dim3(32), dim3(32), 0, stream,
                     whb, zxp, hbuf, ctr, outp);
}
